// PNetwork_57114475102625
// MI455X (gfx1250) — matrix operand order repaired and checked by reading, not compiled, not run on hardware
//
#include <hip/hip_runtime.h>
#include <hip/hip_bf16.h>
#include <stdint.h>

// ---------------------------------------------------------------------------
// Problem constants (B=128, T=256, H=1024)
// ---------------------------------------------------------------------------
#define BQ 128
#define TQ 256
#define HQ 1024
#define K3 3072          // 3*H : concat([x, h, c]) feature dim
#define N3 3072          // zc (H) + gates (2H) output columns
#define NT3 (N3 / 16)    // 192 column tiles in combined weight
#define NTO (HQ / 16)    // 64 column tiles in W_out

typedef __bf16 bf16;
typedef __attribute__((ext_vector_type(16))) __bf16 v16bf;
typedef __attribute__((ext_vector_type(8)))  __bf16 v8bf;
typedef __attribute__((ext_vector_type(8)))  float  v8f;

union AFrag { v16bf v; v8bf h[2]; };

// ---------------------------------------------------------------------------
// Weight prep: build combined 3072x3072 bf16 weight, pre-swizzled into WMMA
// B-matrix fragment order: frag(kt,nt) is 32 lanes x 16 bf16 contiguous.
//   rows k<H      : [ W_tc(x-part) | W_ih ]
//   rows H..2H-1  : [ W_tc(h-part) | W_hh ]
//   rows 2H..3H-1 : [ W_tc(c-part) | 0    ]
// B-fragment layout (32x16 K x N): lanes 0-15 -> N=lane, K=0..15 ;
//                                  lanes 16-31 -> N=lane-16, K=16..31.
// ---------------------------------------------------------------------------
__global__ void prep_wcomb(const float* __restrict__ W_tc,
                           const float* __restrict__ W_ih,
                           const float* __restrict__ W_hh,
                           bf16* __restrict__ Wswz) {
  int idx = blockIdx.x * 256 + threadIdx.x;
  if (idx >= K3 * N3) return;
  int k = idx / N3, n = idx % N3;
  float v;
  if (n < HQ)            v = W_tc[(size_t)k * HQ + n];
  else if (k < HQ)       v = W_ih[(size_t)k * (2 * HQ) + (n - HQ)];
  else if (k < 2 * HQ)   v = W_hh[(size_t)(k - HQ) * (2 * HQ) + (n - HQ)];
  else                   v = 0.0f;
  int kt = k >> 5, kl = k & 31, nt = n >> 4, nl = n & 15;
  int lane = nl + (kl & 16);
  int e = kl & 15;
  Wswz[((((size_t)kt * NT3 + nt) * 32 + lane) << 4) + e] = (bf16)v;
}

__global__ void prep_wout(const float* __restrict__ W_out,
                          bf16* __restrict__ Wswz) {
  int idx = blockIdx.x * 256 + threadIdx.x;
  if (idx >= HQ * HQ) return;
  int k = idx / HQ, n = idx % HQ;
  float v = W_out[(size_t)k * HQ + n];
  int kt = k >> 5, kl = k & 31, nt = n >> 4, nl = n & 15;
  int lane = nl + (kl & 16);
  int e = kl & 15;
  Wswz[((((size_t)kt * NTO + nt) * 32 + lane) << 4) + e] = (bf16)v;
}

__global__ void init_state(float* __restrict__ h0, float* __restrict__ c0,
                           float* __restrict__ contrib) {
  int i = blockIdx.x * 256 + threadIdx.x;
  if (i < BQ * HQ)            h0[i] = 0.0f;
  else if (i < 2 * BQ * HQ)   c0[i - BQ * HQ] = 0.0f;
  else if (i == 2 * BQ * HQ)  *contrib = 0.0f;
}

// ---------------------------------------------------------------------------
// Gates GEMM:  gatesAll(128 x 3072) = [x_t | h | c] @ Wcomb  (bf16 WMMA, f32 acc)
// Block: 128 threads = 4 waves; block tile 64(M) x 64(N); wave tile 16 x 64.
// ---------------------------------------------------------------------------
__global__ __launch_bounds__(128)
void gemm_gates(const float* __restrict__ x, int t,
                const float* __restrict__ h, const float* __restrict__ c,
                const bf16* __restrict__ Wswz, float* __restrict__ gates) {
  __shared__ __align__(16) bf16 Atile[64 * 32];   // 64 rows x 32 k (bf16)

  const int tid   = threadIdx.x;
  const int wave  = tid >> 5;
  const int lane  = tid & 31;
  const int mBlk  = blockIdx.y * 64;
  const int nBlk  = blockIdx.x * 64;
  const int khalf = lane >> 4;
  const int rowA  = 16 * wave + (lane & 15);

  v8f acc[4] = {};

  for (int k0 = 0; k0 < K3; k0 += 32) {
    // --- stage A block into LDS with fp32 -> bf16 conversion ---
    const float* base;
    size_t rstride;
    if (k0 < HQ)            { base = x + (size_t)t * HQ + k0;      rstride = (size_t)TQ * HQ; }
    else if (k0 < 2 * HQ)   { base = h + (k0 - HQ);                rstride = HQ; }
    else                    { base = c + (k0 - 2 * HQ);            rstride = HQ; }
#pragma unroll
    for (int i = 0; i < 4; ++i) {
      int f   = tid + i * 128;          // float4 id: 64 rows x 8 float4
      int row = f >> 3;
      int kq  = f & 7;
      float4 v = *(const float4*)(base + (size_t)(mBlk + row) * rstride + kq * 4);
      bf16* d = &Atile[row * 32 + kq * 4];
      d[0] = (bf16)v.x; d[1] = (bf16)v.y; d[2] = (bf16)v.z; d[3] = (bf16)v.w;
    }
    __syncthreads();

    // --- A fragment: lanes 0-15 rows, khalf selects K {0..15} vs {16..31}
    AFrag af;
    af.h[0] = *(const v8bf*)&Atile[rowA * 32 + khalf * 16];
    af.h[1] = *(const v8bf*)&Atile[rowA * 32 + khalf * 16 + 8];

    const int kt = k0 >> 5;
#pragma unroll
    for (int j = 0; j < 4; ++j) {
      int nt = (nBlk >> 4) + j;
      v16bf bfrag = *(const v16bf*)(Wswz + ((((size_t)kt * NT3 + nt) * 32 + lane) << 4));
      acc[j] = __builtin_amdgcn_wmma_f32_16x16x32_bf16(
          false, af.v, false, bfrag, (short)0, acc[j], false, false);
    }
    __syncthreads();
  }

  // --- epilogue: D layout -> global (lane<16: M=r, lane>=16: M=r+8) ---
  const int mBase = mBlk + 16 * wave + 8 * khalf;
  const int nLane = lane & 15;
#pragma unroll
  for (int j = 0; j < 4; ++j) {
    int n = nBlk + 16 * j + nLane;
#pragma unroll
    for (int r = 0; r < 8; ++r)
      gates[(size_t)(mBase + r) * N3 + n] = acc[j][r];
  }
}

// ---------------------------------------------------------------------------
// Pointwise LSTM cell: zc, g, o, c1, h1 ; emit bf16 h1 for the output GEMM.
// ---------------------------------------------------------------------------
__global__ __launch_bounds__(256)
void pointwise(const float* __restrict__ gates, const float* __restrict__ z, int t,
               const float* __restrict__ c_in,
               const float* __restrict__ b_tc, const float* __restrict__ b_ih,
               const float* __restrict__ b_hh,
               float* __restrict__ h_out, float* __restrict__ c_out,
               bf16* __restrict__ h1bf, float* __restrict__ zc_out) {
  int i = blockIdx.x * 256 + threadIdx.x;      // 0 .. B*H-1
  if (i >= BQ * HQ) return;
  int b = i >> 10, j = i & 1023;

  float zc_pre = gates[(size_t)b * N3 + j] + b_tc[j];
  float g = gates[(size_t)b * N3 + HQ + j] + b_ih[j] + b_hh[j];
  float o = gates[(size_t)b * N3 + 2 * HQ + j] + b_ih[HQ + j] + b_hh[HQ + j];

  zc_pre = fmaxf(zc_pre, 0.0f);
  float zc = 1.0f / (1.0f + __expf(-zc_pre));
  g = tanhf(g);
  o = 1.0f / (1.0f + __expf(-o));

  float ig = z[(((size_t)b * TQ + t) * 2 + 0) * HQ + j];
  float fg = z[(((size_t)b * TQ + t) * 2 + 1) * HQ + j];
  float c1 = fg * c_in[i] + ig * g;
  float h1 = o * tanhf(c1);

  c_out[i] = c1;
  h_out[i] = h1;
  h1bf[i]  = (bf16)h1;
  zc_out[i] = zc;
}

// ---------------------------------------------------------------------------
// Output GEMM: y = relu(h1 @ W_out + b_out); contrib += sum(log(log(zc)+log(y)))
// Same tiling as gemm_gates; A source already bf16.
// ---------------------------------------------------------------------------
__global__ __launch_bounds__(128)
void gemm_out(const bf16* __restrict__ h1, const bf16* __restrict__ Wswz,
              const float* __restrict__ b_out, const float* __restrict__ zc,
              float* __restrict__ contrib) {
  __shared__ __align__(16) bf16 Atile[64 * 32];

  const int tid   = threadIdx.x;
  const int wave  = tid >> 5;
  const int lane  = tid & 31;
  const int mBlk  = blockIdx.y * 64;
  const int nBlk  = blockIdx.x * 64;
  const int khalf = lane >> 4;
  const int rowA  = 16 * wave + (lane & 15);

  v8f acc[4] = {};

  for (int k0 = 0; k0 < HQ; k0 += 32) {
#pragma unroll
    for (int i = 0; i < 2; ++i) {
      int u   = tid + i * 128;          // uint4 id: 64 rows x 4 x 16B
      int row = u >> 2;
      int prt = u & 3;
      *(uint4*)&Atile[row * 32 + prt * 8] =
          *(const uint4*)(h1 + (size_t)(mBlk + row) * HQ + k0 + prt * 8);
    }
    __syncthreads();

    AFrag af;
    af.h[0] = *(const v8bf*)&Atile[rowA * 32 + khalf * 16];
    af.h[1] = *(const v8bf*)&Atile[rowA * 32 + khalf * 16 + 8];

    const int kt = k0 >> 5;
#pragma unroll
    for (int j = 0; j < 4; ++j) {
      int nt = (nBlk >> 4) + j;
      v16bf bfrag = *(const v16bf*)(Wswz + ((((size_t)kt * NTO + nt) * 32 + lane) << 4));
      acc[j] = __builtin_amdgcn_wmma_f32_16x16x32_bf16(
          false, af.v, false, bfrag, (short)0, acc[j], false, false);
    }
    __syncthreads();
  }

  // --- epilogue: relu + bias, fused log-log reduction ---
  const int mBase = mBlk + 16 * wave + 8 * khalf;
  const int nLane = lane & 15;
  float lsum = 0.0f;
#pragma unroll
  for (int j = 0; j < 4; ++j) {
    int n = nBlk + 16 * j + nLane;
    float bo = b_out[n];
#pragma unroll
    for (int r = 0; r < 8; ++r) {
      int m = mBase + r;
      float y = fmaxf(acc[j][r] + bo, 0.0f);
      float lz = __logf(zc[(size_t)m * HQ + n]);
      lsum += __logf(lz + __logf(y));
    }
  }
  atomicAdd(contrib, lsum);
}

// ---------------------------------------------------------------------------
// Pack outputs: [contrib(1), h_n(128*1024), c_n(128*1024)]
// ---------------------------------------------------------------------------
__global__ void pack_out(const float* __restrict__ contrib,
                         const float* __restrict__ hN,
                         const float* __restrict__ cN,
                         float* __restrict__ out) {
  int i = blockIdx.x * 256 + threadIdx.x;
  if (i == 0) out[0] = *contrib;
  if (i < BQ * HQ)            out[1 + i] = hN[i];
  else if (i < 2 * BQ * HQ)   out[1 + i] = cN[i - BQ * HQ];
}

// ---------------------------------------------------------------------------
// Host-side orchestration
// ---------------------------------------------------------------------------
extern "C" void kernel_launch(void* const* d_in, const int* in_sizes, int n_in,
                              void* d_out, int out_size, void* d_ws, size_t ws_size,
                              hipStream_t stream) {
  const float* x    = (const float*)d_in[0];
  // d_in[1] = y (unused by reference)
  const float* z    = (const float*)d_in[2];
  const float* W_tc = (const float*)d_in[3];
  const float* b_tc = (const float*)d_in[4];
  const float* W_ih = (const float*)d_in[5];
  const float* b_ih = (const float*)d_in[6];
  const float* W_hh = (const float*)d_in[7];
  const float* b_hh = (const float*)d_in[8];
  const float* W_out= (const float*)d_in[9];
  const float* b_out= (const float*)d_in[10];

  uint8_t* ws = (uint8_t*)d_ws;
  size_t off = 0;
  bf16*  Wcomb = (bf16*)(ws + off);  off += (size_t)K3 * N3 * 2;        // 18.9 MB
  bf16*  Wout  = (bf16*)(ws + off);  off += (size_t)HQ * HQ * 2;        //  2.1 MB
  float* hbuf0 = (float*)(ws + off); off += (size_t)BQ * HQ * 4;
  float* hbuf1 = (float*)(ws + off); off += (size_t)BQ * HQ * 4;
  float* cbuf0 = (float*)(ws + off); off += (size_t)BQ * HQ * 4;
  float* cbuf1 = (float*)(ws + off); off += (size_t)BQ * HQ * 4;
  float* gates = (float*)(ws + off); off += (size_t)BQ * N3 * 4;
  float* zc    = (float*)(ws + off); off += (size_t)BQ * HQ * 4;
  bf16*  h1bf  = (bf16*)(ws + off);  off += (size_t)BQ * HQ * 2;
  float* contrib = (float*)(ws + off);

  // one-time prep (re-done every call for determinism)
  prep_wcomb<<<(K3 * N3 + 255) / 256, 256, 0, stream>>>(W_tc, W_ih, W_hh, Wcomb);
  prep_wout <<<(HQ * HQ + 255) / 256, 256, 0, stream>>>(W_out, Wout);
  init_state<<<(2 * BQ * HQ + 256) / 256, 256, 0, stream>>>(hbuf0, cbuf0, contrib);

  for (int t = 0; t < TQ; ++t) {
    float* h_cur = (t & 1) ? hbuf1 : hbuf0;
    float* c_cur = (t & 1) ? cbuf1 : cbuf0;
    float* h_nxt = (t & 1) ? hbuf0 : hbuf1;
    float* c_nxt = (t & 1) ? cbuf0 : cbuf1;

    gemm_gates<<<dim3(N3 / 64, BQ / 64), 128, 0, stream>>>(x, t, h_cur, c_cur, Wcomb, gates);
    pointwise <<<(BQ * HQ + 255) / 256, 256, 0, stream>>>(gates, z, t, c_cur,
                                                          b_tc, b_ih, b_hh,
                                                          h_nxt, c_nxt, h1bf, zc);
    gemm_out  <<<dim3(HQ / 64, BQ / 64), 128, 0, stream>>>(h1bf, Wout, b_out, zc, contrib);
  }

  // after t=255 the "next" buffers are hbuf0/cbuf0
  pack_out<<<(2 * BQ * HQ + 255) / 256, 256, 0, stream>>>(contrib, hbuf0, cbuf0, (float*)d_out);
}
